// FeatureVolume_21088289423655
// MI455X (gfx1250) — matrix operand order repaired and checked by reading, not compiled, not run on hardware
//
#include <hip/hip_runtime.h>
#include <stdint.h>

typedef __attribute__((ext_vector_type(16))) _Float16 v16h;
typedef __attribute__((ext_vector_type(8)))  float    v8f;

#define NVIEW 8
#define CIN   32
#define FH    128
#define FW    128

// ---------------------------------------------------------------------------
// Pack W1/W2/W3 into the exact WMMA B-operand lane layout (f16):
//   slot s, lane L: 16 halves = B register contents for that lane.
//   s=0: W1 cols 0..15   s=1: W1 cols 16..31   s=2: W2   s=3: W3 (zero-padded)
// 4 slots * 32 lanes * 16 halves * 2B = 4 KB in d_ws.
// ---------------------------------------------------------------------------
__global__ __launch_bounds__(128)
void fv_pack_weights(const float* __restrict__ W1,
                     const float* __restrict__ W2,
                     const float* __restrict__ W3,
                     _Float16* __restrict__ wpack)
{
    const int t   = threadIdx.x;   // 0..127
    const int L   = t & 31;
    const int s   = t >> 5;        // 0..3
    const int n   = L & 15;
    const int hi  = L >> 4;
    const int klo = hi * 16;
    _Float16* dst = wpack + (size_t)(s * 32 + L) * 16;
#pragma unroll
    for (int q = 0; q < 16; ++q) {
        float v;
        if      (s == 0) v = W1[(klo + q) * 32 + n];
        else if (s == 1) v = W1[(klo + q) * 32 + 16 + n];
        else if (s == 2) v = W2[(klo + q) * 16 + n];
        else             v = (hi == 0 && n < 8) ? W3[q * 8 + n] : 0.f; // K>=16 / n>=8 pad
        dst[q] = (_Float16)v;
    }
}

// ---------------------------------------------------------------------------
// Fused: project + bilinear sample + 3-layer MLP (WMMA) + masked mean/var.
// Block = 256 threads = 8 wave32; wave == view; 16 voxels (consecutive i) per block.
// ---------------------------------------------------------------------------
__global__ __launch_bounds__(256)
void fv_fused_kernel(const float* __restrict__ feats,      // (8,32,128,128)
                     const float* __restrict__ ref_poses,  // (8,3,4)
                     const float* __restrict__ Ks,         // (8,3,3)
                     const float* __restrict__ bbox,       // (2,3)
                     const int*   __restrict__ img_h_p,
                     const int*   __restrict__ img_w_p,
                     const float* __restrict__ b1,
                     const float* __restrict__ b2,
                     const float* __restrict__ b3,
                     const _Float16* __restrict__ wpack,   // 4KB packed B operands
                     float* __restrict__ out)              // (1,16,64,64,64)
{
    __shared__ _Float16 wlds[4 * 32 * 16];  // 4 KB packed weights (shared by all 8 waves)
    __shared__ float fbuf[NVIEW][16][CIN];  // 16 KB per-wave scratch: feats -> h1 -> h2
    __shared__ float wbuf[NVIEW][16];       // per-view per-voxel mask
    __shared__ float invs[16];              // 1/(sum mask + 1e-8)
    __shared__ float acc1[16][8];           // weighted mean
    __shared__ float acc2[16][8];           // weighted variance

    const int tid  = threadIdx.x;
    const int wv   = tid >> 5;      // wave id == view id (8 waves)
    const int lane = tid & 31;
    const int m    = lane & 15;     // matrix row (voxel in tile)
    const int hi   = lane >> 4;     // lane half
    const int n    = m;             // B/C/D column for this lane

    if (tid < 128) { acc1[tid >> 3][tid & 7] = 0.f; acc2[tid >> 3][tid & 7] = 0.f; }

    // ---- async broadcast of packed weights into LDS: each wave moves 512 B ----
    {
        const unsigned off = (unsigned)(wv * 512 + lane * 16);
        const char*    gsrc   = (const char*)wpack + off;
        const unsigned ldsoff = (unsigned)(uintptr_t)((char*)wlds + off); // low 32b = LDS byte addr
        asm volatile("global_load_async_to_lds_b128 %0, %1, off\n\t"
                     "s_wait_asynccnt 0"
                     :: "v"(ldsoff), "v"(gsrc) : "memory");
    }

    // block -> (i-tile, j, k)
    const int blk = blockIdx.x;
    const int k   = blk & 63;
    const int j   = (blk >> 6) & 63;
    const int i0  = (blk >> 12) << 4;

    const float VOXEL = 0.3f / 64.f;
    const float HALFV = VOXEL * 0.5f;

    // ---- P = K @ [R|t] for this wave's view ----
    float P[3][4];
    {
        const float* Kv = Ks + wv * 9;
        const float* Pw = ref_poses + wv * 12;
#pragma unroll
        for (int r = 0; r < 3; ++r)
#pragma unroll
            for (int c4 = 0; c4 < 4; ++c4) {
                float s = 0.f;
#pragma unroll
                for (int q = 0; q < 3; ++q) s += Kv[r * 3 + q] * Pw[q * 4 + c4];
                P[r][c4] = s;
            }
    }
    const float img_w = (float)img_w_p[0];
    const float img_h = (float)img_h_p[0];
    const float b1a = b1[n], b1b = b1[16 + n], b2n = b2[n];
    const float b3n = (n < 8) ? b3[n] : 0.f;

    // ---- project voxel m, bilinear-sample 16 channels per lane ----
    const float px = (float)(i0 + m) * VOXEL + HALFV + bbox[0];
    const float py = (float)j        * VOXEL + HALFV + bbox[1];
    const float pz = (float)k        * VOXEL + HALFV + bbox[2];

    const float u  = P[0][0]*px + P[0][1]*py + P[0][2]*pz + P[0][3];
    const float vv = P[1][0]*px + P[1][1]*py + P[1][2]*pz + P[1][3];
    const float z  = P[2][0]*px + P[2][1]*py + P[2][2]*pz + P[2][3];
    const float rz = (z > 0.f) ? (1.f / z) : 0.f;   // finite when behind camera
    const float ix = (u  * rz / img_w) * 0.5f * (float)(FW - 1);
    const float iy = (vv * rz / img_h) * 0.5f * (float)(FH - 1);
    const float ix0 = floorf(ix), iy0 = floorf(iy);
    const float ix1 = ix0 + 1.f,  iy1 = iy0 + 1.f;
    const float w_nw = (ix1 - ix) * (iy1 - iy);
    const float w_ne = (ix - ix0) * (iy1 - iy);
    const float w_sw = (ix1 - ix) * (iy - iy0);
    const float w_se = (ix - ix0) * (iy - iy0);
    const int cx0 = (int)fminf(fmaxf(ix0, 0.f), (float)(FW - 1));
    const int cx1 = (int)fminf(fmaxf(ix1, 0.f), (float)(FW - 1));
    const int cy0 = (int)fminf(fmaxf(iy0, 0.f), (float)(FH - 1));
    const int cy1 = (int)fminf(fmaxf(iy1, 0.f), (float)(FH - 1));
    const float maskv = (ix >= 0.f && ix <= (float)(FW - 1) &&
                         iy >= 0.f && iy <= (float)(FH - 1) && z > 0.f) ? 1.f : 0.f;

    const int cbase = hi * 16;
#pragma unroll
    for (int cc = 0; cc < 16; ++cc) {
        const int c = cbase + cc;
        const float* fp = feats + (size_t)(wv * CIN + c) * (FH * FW);
        const float vnw = fp[cy0 * FW + cx0];
        const float vne = fp[cy0 * FW + cx1];
        const float vsw = fp[cy1 * FW + cx0];
        const float vse = fp[cy1 * FW + cx1];
        fbuf[wv][m][c] = vnw * w_nw + vne * w_ne + vsw * w_sw + vse * w_se;
    }
    if (hi == 0) wbuf[wv][m] = maskv;

    __syncthreads();   // packed weights in LDS + masks + zeroed accumulators visible

    // ---- B operands: two contiguous ds_load_b128 per matrix from shared LDS copy ----
    const v16h B1a = *(const v16h*)(wlds + (0 * 32 + lane) * 16);
    const v16h B1b = *(const v16h*)(wlds + (1 * 32 + lane) * 16);
    const v16h B2v = *(const v16h*)(wlds + (2 * 32 + lane) * 16);
    const v16h B3v = *(const v16h*)(wlds + (3 * 32 + lane) * 16);

    // ---- A1: 16x32 f16, A layout (lane row m; K contiguous hi*16..hi*16+15) ----
    const int koff = hi * 8;
    v16h A1;
#pragma unroll
    for (int q = 0; q < 16; ++q) {
        const int c = hi * 16 + q;
        A1[q] = (_Float16)fbuf[wv][m][c];
    }

    v8f zero = {};
    // layer 1: 32 -> 32 (two N tiles)
    v8f c1a = __builtin_amdgcn_wmma_f32_16x16x32_f16(false, A1, false, B1a, (short)0, zero, false, false);
    v8f c1b = __builtin_amdgcn_wmma_f32_16x16x32_f16(false, A1, false, B1b, (short)0, zero, false, false);

#pragma unroll
    for (int idx = 0; idx < 8; ++idx) {       // bias + relu, transpose via per-wave LDS
        const int row = idx + 8 * hi;
        fbuf[wv][row][n]      = fmaxf(c1a[idx] + b1a, 0.f);
        fbuf[wv][row][16 + n] = fmaxf(c1b[idx] + b1b, 0.f);
    }

    v16h A2;
#pragma unroll
    for (int q = 0; q < 16; ++q) {
        const int c = hi * 16 + q;
        A2[q] = (_Float16)fbuf[wv][m][c];
    }
    // layer 2: 32 -> 16
    v8f c2 = __builtin_amdgcn_wmma_f32_16x16x32_f16(false, A2, false, B2v, (short)0, zero, false, false);

#pragma unroll
    for (int idx = 0; idx < 8; ++idx) {
        const int row = idx + 8 * hi;
        fbuf[wv][row][n] = fmaxf(c2[idx] + b2n, 0.f);   // h2: 16x16
    }
    v16h A3;
#pragma unroll
    for (int q = 0; q < 16; ++q)
        A3[q] = (hi == 0) ? (_Float16)fbuf[wv][m][q] : (_Float16)0.f;  // K padded 16->32
    // layer 3: 16 -> 8 (N padded to 16 with zero B columns)
    v8f c3 = __builtin_amdgcn_wmma_f32_16x16x32_f16(false, A3, false, B3v, (short)0, zero, false, false);

    // per-voxel weight normalizer
    if (tid < 16) {
        float s = 0.f;
#pragma unroll
        for (int q = 0; q < NVIEW; ++q) s += wbuf[q][tid];
        invs[tid] = 1.f / (s + 1e-8f);
    }
    __syncthreads();

    // pass 1: weighted mean across the 8 waves (views) via LDS float atomics
    if (n < 8) {
#pragma unroll
        for (int idx = 0; idx < 8; ++idx) {
            const int row = idx + 8 * hi;
            const float w = wbuf[wv][row] * invs[row];
            atomicAdd(&acc1[row][n], w * (c3[idx] + b3n));
        }
    }
    __syncthreads();

    // pass 2: weighted variance around the mean (matches reference exactly)
    if (n < 8) {
#pragma unroll
        for (int idx = 0; idx < 8; ++idx) {
            const int row = idx + 8 * hi;
            const float w = wbuf[wv][row] * invs[row];
            const float d = (c3[idx] + b3n) - acc1[row][n];
            atomicAdd(&acc2[row][n], w * d * d);
        }
    }
    __syncthreads();

    // out[0][c][k][j][i] = mean ; out[0][c+8][k][j][i] = var ; coalesced along i
    if (tid < 128) {
        const int c  = tid >> 4;     // 0..7
        const int mm = tid & 15;
        const size_t sp = (size_t)k * 4096 + (size_t)j * 64 + (size_t)(i0 + mm);
        out[(size_t)c       * 262144 + sp] = acc1[mm][c];
        out[(size_t)(c + 8) * 262144 + sp] = acc2[mm][c];
    }
}

extern "C" void kernel_launch(void* const* d_in, const int* in_sizes, int n_in,
                              void* d_out, int out_size, void* d_ws, size_t ws_size,
                              hipStream_t stream) {
    const float* feats     = (const float*)d_in[0];
    const float* ref_poses = (const float*)d_in[1];
    const float* Ks        = (const float*)d_in[2];
    const float* bbox      = (const float*)d_in[3];
    const int*   img_h     = (const int*)  d_in[4];
    const int*   img_w     = (const int*)  d_in[5];
    const float* W1        = (const float*)d_in[6];
    const float* b1        = (const float*)d_in[7];
    const float* W2        = (const float*)d_in[8];
    const float* b2        = (const float*)d_in[9];
    const float* W3        = (const float*)d_in[10];
    const float* b3        = (const float*)d_in[11];

    _Float16* wpack = (_Float16*)d_ws;   // 4 KB packed WMMA-B weights

    fv_pack_weights<<<1, 128, 0, stream>>>(W1, W2, W3, wpack);

    // 64^3 voxels / 16 per block = 16384 blocks, 256 threads (8 wave32 = 8 views)
    fv_fused_kernel<<<16384, 256, 0, stream>>>(feats, ref_poses, Ks, bbox,
                                               img_h, img_w,
                                               b1, b2, b3, wpack,
                                               (float*)d_out);
}
